// LSTM_block_69277822485312
// MI455X (gfx1250) — matrix operand order repaired and checked by reading, not compiled, not run on hardware
//
#include <hip/hip_runtime.h>
#include <hip/hip_bf16.h>

typedef __bf16 bf16;
typedef __attribute__((ext_vector_type(16))) __bf16 v16bf;
typedef __attribute__((ext_vector_type(8)))  __bf16 v8bf;
typedef __attribute__((ext_vector_type(8)))  float  v8f;
typedef __attribute__((ext_vector_type(4)))  unsigned u32x4;
typedef __attribute__((ext_vector_type(8)))  int      i32x8;
typedef __attribute__((ext_vector_type(4)))  int      i32x4;

#define AS3 __attribute__((address_space(3)))
typedef AS3 __bf16   lds_bf16;
typedef AS3 v8bf     lds_v8bf;
typedef AS3 unsigned lds_u32;

#define DI __device__ __forceinline__

constexpr int B_  = 64;
constexpr int T_  = 256;
constexpr int KIN = 5;
constexpr int X_  = 512;
constexpr int H_  = 512;
constexpr int G_  = 4 * H_;   // 2048 gate width
constexpr int F_  = 2 * X_;   // 1024 input feature width
constexpr int M_  = B_ * T_;  // 16384 GEMM rows

// ---------------- WMMA fragment helpers (bf16 16x16x32, wave32) ----------------
// A: 16x32 row-major. lanes 0-15 row m: e0..7=K0..7, e8..15=K16..23;
// lanes 16-31 row m: e0..7=K8..15, e8..15=K24..31.
DI v16bf load_a_frag(const bf16* A, int lda, int m_base, int k0, int lane) {
  int m    = m_base + (lane & 15);
  int half = lane >> 4;
  const bf16* row = A + (size_t)m * lda + k0;
  v8bf lo = *(const v8bf*)(row + 8 * half);
  v8bf hi = *(const v8bf*)(row + 16 + 8 * half);
  v16bf r;
#pragma unroll
  for (int i = 0; i < 8; ++i) { r[i] = lo[i]; r[8 + i] = hi[i]; }
  return r;
}

// B: 32x16, supplied as Bt row-major [n][k] (= weight matrix W[g][f] as stored).
DI v16bf load_b_frag(const bf16* Bt, int ldb, int n_base, int k0, int lane) {
  int n    = n_base + (lane & 15);
  int half = lane >> 4;
  const bf16* row = Bt + (size_t)n * ldb + k0;
  v8bf lo = *(const v8bf*)(row + 8 * half);
  v8bf hi = *(const v8bf*)(row + 16 + 8 * half);
  v16bf r;
#pragma unroll
  for (int i = 0; i < 8; ++i) { r[i] = lo[i]; r[8 + i] = hi[i]; }
  return r;
}

// ---- LDS (address_space(3)) variants: guaranteed ds_load_b128 path ----
DI v16bf load_a_frag_lds(const lds_bf16* A, int lda, int k0, int lane) {
  int m    = lane & 15;
  int half = lane >> 4;
  const lds_bf16* row = A + m * lda + k0;
  v8bf lo = *(const lds_v8bf*)(row + 8 * half);
  v8bf hi = *(const lds_v8bf*)(row + 16 + 8 * half);
  v16bf r;
#pragma unroll
  for (int i = 0; i < 8; ++i) { r[i] = lo[i]; r[8 + i] = hi[i]; }
  return r;
}

DI v16bf load_b_frag_lds(const lds_bf16* Bt, int ldb, int k0, int lane) {
  int n    = lane & 15;
  int half = lane >> 4;
  const lds_bf16* row = Bt + n * ldb + k0;
  v8bf lo = *(const lds_v8bf*)(row + 8 * half);
  v8bf hi = *(const lds_v8bf*)(row + 16 + 8 * half);
  v16bf r;
#pragma unroll
  for (int i = 0; i < 8; ++i) { r[i] = lo[i]; r[8 + i] = hi[i]; }
  return r;
}

DI v8f wmma_bf16(v16bf a, v16bf b, v8f c) {
  return __builtin_amdgcn_wmma_f32_16x16x32_bf16(false, a, false, b, (short)0, c,
                                                 false, false);
}

DI float sigmoidf_(float x) { return 1.0f / (1.0f + __expf(-x)); }

// ---------------- TDM: 2D tile (4-byte elements) global -> LDS ----------------
// D# layout per CDNA5 ISA; 6-arg builtin on this toolchain.
DI void tdm_load_2d_to_lds(unsigned lds_byte_off, const void* gaddr,
                           unsigned row_bytes, unsigned rows, unsigned stride_bytes) {
  unsigned u_row = row_bytes >> 2;      // tensor_dim0 / tile_dim0 in 4B units
  unsigned u_str = stride_bytes >> 2;   // tensor_dim0_stride in 4B units
  unsigned long long ga = (unsigned long long)(uintptr_t)gaddr;
  u32x4 g0;
  g0[0] = 1u;                                                  // count=1, user mode
  g0[1] = lds_byte_off;                                        // lds_addr
  g0[2] = (unsigned)(ga & 0xffffffffu);                        // global_addr lo
  g0[3] = (unsigned)((ga >> 32) & 0x01ffffffu) | (2u << 30);   // global_addr hi | type=2
  i32x8 g1;
  g1[0] = (int)(2u << 16);                                     // data_size = 4B
  g1[1] = (int)((u_row & 0xffffu) << 16);                      // tensor_dim0[15:0]
  g1[2] = (int)((u_row >> 16) | ((rows & 0xffffu) << 16));     // dim0[31:16] | dim1[15:0]
  g1[3] = (int)((rows >> 16) | ((u_row & 0xffffu) << 16));     // dim1[31:16] | tile_dim0
  g1[4] = (int)(rows & 0xffffu);                               // tile_dim1 (tile_dim2=0)
  g1[5] = (int)u_str;                                          // dim0_stride[31:0]
  g1[6] = 0;                                                   // dim0_stride[47:32]
  g1[7] = 0;                                                   // dim1_stride unused (2D)
  i32x4 z4;
  z4[0] = 0; z4[1] = 0; z4[2] = 0; z4[3] = 0;
  i32x8 z8;
  z8[0] = 0; z8[1] = 0; z8[2] = 0; z8[3] = 0;
  z8[4] = 0; z8[5] = 0; z8[6] = 0; z8[7] = 0;
  __builtin_amdgcn_tensor_load_to_lds(g0, g1, z4, z4, z8, 0);
}

// ---------------- device-wide sense-reversing barrier ----------------
DI void grid_barrier(unsigned* cnt, unsigned* gen, unsigned nb) {
  __threadfence();
  __syncthreads();
  if (threadIdx.x == 0) {
    unsigned g = __hip_atomic_load(gen, __ATOMIC_ACQUIRE, __HIP_MEMORY_SCOPE_AGENT);
    unsigned a = __hip_atomic_fetch_add(cnt, 1u, __ATOMIC_ACQ_REL, __HIP_MEMORY_SCOPE_AGENT);
    if (a == nb - 1u) {
      __hip_atomic_store(cnt, 0u, __ATOMIC_RELAXED, __HIP_MEMORY_SCOPE_AGENT);
      __hip_atomic_fetch_add(gen, 1u, __ATOMIC_RELEASE, __HIP_MEMORY_SCOPE_AGENT);
    } else {
      while (__hip_atomic_load(gen, __ATOMIC_ACQUIRE, __HIP_MEMORY_SCOPE_AGENT) == g)
        __builtin_amdgcn_s_sleep(2);
    }
  }
  __syncthreads();
}

// ---------------- prep: bf16 weights, fused bias, zero state ----------------
__global__ void k_prep(const float* __restrict__ Wih_f, const float* __restrict__ Whh_f,
                       const float* __restrict__ bih, const float* __restrict__ bhh,
                       bf16* __restrict__ Wih, bf16* __restrict__ Whh,
                       float* __restrict__ bias, float* __restrict__ c,
                       bf16* __restrict__ hbf, unsigned* __restrict__ sync) {
  size_t tid    = (size_t)blockIdx.x * blockDim.x + threadIdx.x;
  size_t stride = (size_t)gridDim.x * blockDim.x;
  for (size_t i = tid; i < (size_t)G_ * F_; i += stride) Wih[i] = (bf16)Wih_f[i];
  for (size_t i = tid; i < (size_t)G_ * H_; i += stride) Whh[i] = (bf16)Whh_f[i];
  for (size_t i = tid; i < (size_t)G_; i += stride)      bias[i] = bih[i] + bhh[i];
  for (size_t i = tid; i < (size_t)B_ * H_; i += stride) { c[i] = 0.0f; hbf[i] = (bf16)0.0f; }
  if (tid < 8) sync[tid] = 0u;
}

// ---------------- build xg = concat(x[:,:,0,:], mean(x[:,:,1:5,:])) in bf16 ----------------
__global__ void k_xg(const float* __restrict__ x, bf16* __restrict__ xg) {
  size_t tid    = (size_t)blockIdx.x * blockDim.x + threadIdx.x;
  size_t stride = (size_t)gridDim.x * blockDim.x;
  for (size_t i = tid; i < (size_t)M_ * F_; i += stride) {
    size_t m = i >> 10;             // row (b*T + t)
    int    f = (int)(i & (F_ - 1));
    const float* base = x + m * (size_t)(KIN * X_);
    float v;
    if (f < X_) {
      v = base[f];
    } else {
      int fm = f - X_;
      v = 0.25f * (base[X_ + fm] + base[2 * X_ + fm] + base[3 * X_ + fm] + base[4 * X_ + fm]);
    }
    xg[i] = (bf16)v;
  }
}

// ---------------- big GEMM: pre[t,b,g] = xg[m] . Wih[g] + bias[g] ----------------
__global__ void k_gemm_pre(const bf16* __restrict__ xg, const bf16* __restrict__ Wih,
                           const float* __restrict__ bias, float* __restrict__ pre) {
  int lane = threadIdx.x & 31;
  int wid  = blockIdx.x * (blockDim.x >> 5) + (threadIdx.x >> 5); // 0..32767
  int ng   = wid & 31;     // n-group of 64 cols
  int mt   = wid >> 5;     // m-tile
  int n0   = ng * 64;
  int half = lane >> 4, n = lane & 15;

  v8f acc[4];
#pragma unroll
  for (int s = 0; s < 4; ++s) {
    float bv = bias[n0 + s * 16 + n];
#pragma unroll
    for (int r = 0; r < 8; ++r) acc[s][r] = bv;
  }
  for (int k0 = 0; k0 < F_; k0 += 32) {
    v16bf a = load_a_frag(xg, F_, mt * 16, k0, lane);
#pragma unroll
    for (int s = 0; s < 4; ++s) {
      v16bf b = load_b_frag(Wih, F_, n0 + s * 16, k0, lane);
      acc[s] = wmma_bf16(a, b, acc[s]);
    }
  }
#pragma unroll
  for (int s = 0; s < 4; ++s) {
#pragma unroll
    for (int r = 0; r < 8; ++r) {
      int Mrow = mt * 16 + 8 * half + r;
      int b = Mrow >> 8, t = Mrow & (T_ - 1);
      pre[((size_t)(t * B_ + b)) * G_ + n0 + s * 16 + n] = acc[s][r];
    }
  }
}

// ---------------- persistent LSTM scan: 64 blocks x 256 threads = 512 waves ----------------
// LDS: per-wave Whh slice (16 gate rows x 512, staged ONCE via TDM) + per-block
// h tile (16 batch rows x 512, staged via TDM each step; all 8 waves share mt).
struct alignas(16) ScanLds {
  bf16 whh[8][16 * H_];   // 128 KB
  bf16 h[16 * H_];        //  16 KB
};
constexpr unsigned WHH_WAVE_SZ = 16 * H_ * sizeof(bf16);   // 16 KB
constexpr unsigned H_REL_OFF   = 8 * WHH_WAVE_SZ;          // 131072 within struct

__global__ void k_scan(const float* __restrict__ pre, const bf16* __restrict__ Whh,
                       float* __restrict__ z, bf16* __restrict__ hbf,
                       float* __restrict__ c, float* __restrict__ hbuf,
                       unsigned* __restrict__ sync) {
  __shared__ ScanLds sm;
  const unsigned nb = gridDim.x;
  int lane = threadIdx.x & 31;
  int warp = threadIdx.x >> 5;
  int wid  = blockIdx.x * 8 + warp;  // 0..511 == exactly 4x128 tiles of z
  int mt   = wid >> 7;               // 0..3, identical for all 8 waves of a block
  int nt   = wid & 127;              // 0..127 (gate cols 16*nt..)
  int tidg = blockIdx.x * blockDim.x + threadIdx.x; // 0..16383
  int half = lane >> 4, n = lane & 15;

  // Raw LDS byte offset of sm: generic LDS addr keeps the offset in its low 32 bits.
  unsigned sm_base = (unsigned)(uintptr_t)(void*)&sm;
  unsigned whh_off = sm_base + (unsigned)warp * WHH_WAVE_SZ;
  unsigned h_off   = sm_base + H_REL_OFF;
  const lds_bf16* whh_l = (const lds_bf16*)whh_off;  // as(3) inttoptr
  const lds_bf16* h_l   = (const lds_bf16*)h_off;

  // Visible zero-fill of sm so the compiler cannot fold LDS loads of a
  // "never-stored" object to poison (TDM writes are invisible to LLVM).
  {
    lds_u32* smw = (lds_u32*)sm_base;
    for (int i = threadIdx.x; i < (int)(sizeof(ScanLds) / 4); i += 256) smw[i] = 0u;
  }
  __syncthreads();

  // Stage this wave's constant Whh slice into LDS once (TDM, 16 rows x 1KB).
  tdm_load_2d_to_lds(whh_off, Whh + (size_t)nt * 16 * H_,
                     H_ * sizeof(bf16), 16, H_ * sizeof(bf16));
  __builtin_amdgcn_s_wait_tensorcnt(0);
  __syncthreads();
  asm volatile("" ::: "memory");   // LDS contents changed behind the compiler's back

  for (int t = 0; t < T_; ++t) {
    // ---- stage h tile (block-shared A operand) into LDS via TDM ----
    if (warp == 0) {
      tdm_load_2d_to_lds(h_off, hbf + (size_t)mt * 16 * H_,
                         H_ * sizeof(bf16), 16, H_ * sizeof(bf16));
      __builtin_amdgcn_s_wait_tensorcnt(0);
    }
    __syncthreads();
    asm volatile("" ::: "memory"); // h tile rewritten by TDM

    // ---- phase A: z = pre[t] + h @ Whh^T  (WMMA fed from LDS via ds_load) ----
    const float* preT = pre + (size_t)t * B_ * G_;
    v8f acc;
#pragma unroll
    for (int r = 0; r < 8; ++r)
      acc[r] = preT[(size_t)(mt * 16 + 8 * half + r) * G_ + nt * 16 + n];
#pragma unroll 4
    for (int k0 = 0; k0 < H_; k0 += 32) {
      v16bf a = load_a_frag_lds(h_l, H_, k0, lane);
      v16bf b = load_b_frag_lds(whh_l, H_, k0, lane);
      acc = wmma_bf16(a, b, acc);
    }
#pragma unroll
    for (int r = 0; r < 8; ++r)
      z[(size_t)(mt * 16 + 8 * half + r) * G_ + nt * 16 + n] = acc[r];

    grid_barrier(sync, sync + 1, nb);

    // ---- phase B: gates + state update (+ prefetch next step's pre) ----
    if (t + 1 < T_) {
      const float* pn = pre + (size_t)(t + 1) * B_ * G_;
      __builtin_prefetch(pn + (size_t)tidg * 8, 0, 1);
    }
    for (int idx = tidg; idx < B_ * H_; idx += (int)nb * 256) {
      int bb = idx >> 9, j = idx & (H_ - 1);
      const float* zr = z + (size_t)bb * G_;
      float ig = sigmoidf_(zr[j]);
      float fg = sigmoidf_(zr[H_ + j]);
      float gg = tanhf(zr[2 * H_ + j]);
      float og = sigmoidf_(zr[3 * H_ + j]);
      float cn = fg * c[idx] + ig * gg;
      float hn = og * tanhf(cn);
      c[idx] = cn;
      hbuf[(size_t)t * (B_ * H_) + idx] = hn;
      hbf[idx] = (bf16)hn;
    }

    grid_barrier(sync, sync + 1, nb);
  }
}

// ---------------- per-timestep batch-norm statistics ----------------
__global__ void k_stats(const float* __restrict__ hbuf, float* __restrict__ stats) {
  int t = blockIdx.x;
  const float* p = hbuf + (size_t)t * (B_ * H_);
  float s = 0.0f, s2 = 0.0f;
  for (int i = threadIdx.x; i < B_ * H_; i += blockDim.x) {
    float v = p[i]; s += v; s2 += v * v;
  }
  __shared__ float sh0[256], sh1[256];
  sh0[threadIdx.x] = s; sh1[threadIdx.x] = s2;
  __syncthreads();
  for (int off = 128; off > 0; off >>= 1) {
    if ((int)threadIdx.x < off) {
      sh0[threadIdx.x] += sh0[threadIdx.x + off];
      sh1[threadIdx.x] += sh1[threadIdx.x + off];
    }
    __syncthreads();
  }
  if (threadIdx.x == 0) {
    const float inv = 1.0f / (float)(B_ * H_);
    float mean = sh0[0] * inv;
    float var  = sh1[0] * inv - mean * mean;
    stats[2 * t]     = mean;
    stats[2 * t + 1] = rsqrtf(var + 1e-5f);
  }
}

// ---------------- BN apply + ELU + mean over T ----------------
__global__ void k_out(const float* __restrict__ hbuf, const float* __restrict__ stats,
                      const float* __restrict__ gamma, const float* __restrict__ beta,
                      float* __restrict__ out) {
  int idx = blockIdx.x * blockDim.x + threadIdx.x;  // 0..B_*H_-1
  float acc = 0.0f;
  for (int t = 0; t < T_; ++t) {
    float v = hbuf[(size_t)t * (B_ * H_) + idx];
    float y = (v - stats[2 * t]) * stats[2 * t + 1] * gamma[t] + beta[t];
    acc += (y > 0.0f) ? y : (__expf(y) - 1.0f);
  }
  out[idx] = acc * (1.0f / (float)T_);
}

// ---------------- host side ----------------
extern "C" void kernel_launch(void* const* d_in, const int* in_sizes, int n_in,
                              void* d_out, int out_size, void* d_ws, size_t ws_size,
                              hipStream_t stream) {
  const float* x     = (const float*)d_in[0];
  const float* Wih_f = (const float*)d_in[1];
  const float* Whh_f = (const float*)d_in[2];
  const float* bih   = (const float*)d_in[3];
  const float* bhh   = (const float*)d_in[4];
  const float* gamma = (const float*)d_in[5];
  const float* beta  = (const float*)d_in[6];

  char* ws = (char*)d_ws;
  size_t off = 0;
  auto alloc = [&](size_t bytes) -> char* {
    char* p = ws + off;
    off = (off + bytes + 255) & ~(size_t)255;
    return p;
  };

  float* pre      = (float*)alloc((size_t)M_ * G_ * sizeof(float));   // 128 MB
  char*  xg_hbuf  = alloc((size_t)M_ * F_ * sizeof(bf16));            // 32 MB (aliased)
  bf16*  xg       = (bf16*)xg_hbuf;    // live until k_gemm_pre finishes
  float* hbuf     = (float*)xg_hbuf;   // live from k_scan on (same 32 MB)
  bf16*  Wih      = (bf16*)alloc((size_t)G_ * F_ * sizeof(bf16));     // 4 MB
  bf16*  Whh      = (bf16*)alloc((size_t)G_ * H_ * sizeof(bf16));     // 2 MB
  float* bias     = (float*)alloc((size_t)G_ * sizeof(float));
  float* z        = (float*)alloc((size_t)B_ * G_ * sizeof(float));
  bf16*  hbf      = (bf16*)alloc((size_t)B_ * H_ * sizeof(bf16));
  float* c        = (float*)alloc((size_t)B_ * H_ * sizeof(float));
  float* stats    = (float*)alloc((size_t)T_ * 2 * sizeof(float));
  unsigned* sync  = (unsigned*)alloc(256);
  (void)in_sizes; (void)n_in; (void)out_size; (void)ws_size;

  k_prep<<<2048, 256, 0, stream>>>(Wih_f, Whh_f, bih, bhh, Wih, Whh, bias, c, hbf, sync);
  k_xg<<<8192, 256, 0, stream>>>(x, xg);
  k_gemm_pre<<<4096, 256, 0, stream>>>(xg, Wih, bias, pre);   // 32768 wave-tasks exactly
  k_scan<<<64, 256, 0, stream>>>(pre, Whh, z, hbf, c, hbuf, sync); // persistent, resident
  k_stats<<<T_, 256, 0, stream>>>(hbuf, stats);
  k_out<<<(B_ * H_) / 256, 256, 0, stream>>>(hbuf, stats, gamma, beta, (float*)d_out);
}
